// TransformerBlock_188978561654
// MI455X (gfx1250) — matrix operand order repaired and checked by reading, not compiled, not run on hardware
//
#include <hip/hip_runtime.h>
#include <cstdint>

#define D_MODEL 768
#define N_HEADS 12
#define D_HEAD  64
#define D_MLP   3072
#define SEQ     2048
#define BATCH   2
#define ROWS    (BATCH * SEQ)   // 4096

typedef unsigned short u16;
typedef __attribute__((ext_vector_type(16))) __bf16 v16bf;
typedef __attribute__((ext_vector_type(8)))  float  v8f;

union AFrag { uint4 u[2]; v16bf v; };

__device__ __forceinline__ u16 f2bf(float f) {
    unsigned u = __float_as_uint(f);
    u += 0x7FFFu + ((u >> 16) & 1u);          // round-to-nearest-even
    return (u16)(u >> 16);
}

__device__ __forceinline__ v8f v8zero() {
    v8f z;
#pragma unroll
    for (int i = 0; i < 8; ++i) z[i] = 0.0f;
    return z;
}

__device__ __forceinline__ v8f wmma_bf16(v16bf a, v16bf b, v8f c) {
    // D = A(16x32 bf16) * B(32x16 bf16) + C(16x16 f32)
    return __builtin_amdgcn_wmma_f32_16x16x32_bf16(
        /*neg_a=*/false, a, /*neg_b=*/false, b,
        /*c_mod=*/(short)0, c, /*reuse_a=*/false, /*reuse_b=*/false);
}

// LDS byte offset of a pointer into a __shared__ array (AS(3) pointers are
// 32-bit offsets from the workgroup LDS base).
__device__ __forceinline__ unsigned lds_offset(const u16* p) {
    return (unsigned)(size_t)(const __attribute__((address_space(3))) u16*)p;
}

// Async memory -> LDS copy of 16 bytes (GLOBAL_LOAD_ASYNC_TO_LDS_B128,
// tracked by ASYNCcnt; no VGPR data path).
__device__ __forceinline__ void async_copy_b128(const u16* lds_dst, const u16* gsrc) {
    unsigned dst = lds_offset(lds_dst);
    unsigned long long src = (unsigned long long)(const void*)gsrc;
    asm volatile("global_load_async_to_lds_b128 %0, %1, off"
                 :: "v"(dst), "v"(src) : "memory");
}

__device__ __forceinline__ void wait_async_all() {
    asm volatile("s_wait_asynccnt 0x0" ::: "memory");
}

// A fragment (16 rows x 32 K), rows contiguous with leading dim ldk (elements).
// ISA 7.12.2: lanes 0-15 hold M=lane, K={0-7,16-23}; lanes 16-31 hold M=lane-16, K={8-15,24-31}.
__device__ __forceinline__ v16bf load_a_frag(const u16* base, int ldk, int row, int lane) {
    int r   = row + (lane & 15);
    int sel = lane >> 4;
    AFrag f;
    f.u[0] = *(const uint4*)(base + (size_t)r * ldk + sel * 8);
    f.u[1] = *(const uint4*)(base + (size_t)r * ldk + 16 + sel * 8);
    return f.v;
}

// B fragment (32 K x 16 cols), stored column-major: K contiguous per column, leading dim ldk.
// lanes 0-15 hold N=lane, K={0-7,16-23}; lanes 16-31 hold N=lane-16, K={8-15,24-31}.
__device__ __forceinline__ v16bf load_b_frag(const u16* base, int ldk, int col, int lane) {
    int c   = col + (lane & 15);
    int sel = lane >> 4;
    AFrag f;
    f.u[0] = *(const uint4*)(base + (size_t)c * ldk + sel * 8);
    f.u[1] = *(const uint4*)(base + (size_t)c * ldk + 16 + sel * 8);
    return f.v;
}

// ---------------------------------------------------------------------------
// fp32 -> bf16 converters (weights)
// ---------------------------------------------------------------------------
__global__ __launch_bounds__(256) void k_cvt(const float* __restrict__ in,
                                             u16* __restrict__ out, int n) {
    int i = blockIdx.x * 256 + threadIdx.x;
    if (i < n) out[i] = f2bf(in[i]);
}

// W_{Q,K,V}[h][d][e] -> W'[d][h*64+e]  (so q = x @ W' is a plain GEMM)
__global__ __launch_bounds__(256) void k_cvt_qkvw(const float* __restrict__ in,
                                                  u16* __restrict__ out) {
    int i = blockIdx.x * 256 + threadIdx.x;          // over 768*768
    int d = i / D_MODEL;
    int c = i - d * D_MODEL;
    int h = c >> 6, e = c & 63;
    out[i] = f2bf(in[((size_t)h * D_MODEL + d) * D_HEAD + e]);
}

// ---------------------------------------------------------------------------
// LayerNorm (768 cols), one 256-thread block per row, bf16 output
// ---------------------------------------------------------------------------
__global__ __launch_bounds__(256) void k_layernorm(const float* __restrict__ x,
                                                   const float* __restrict__ w,
                                                   const float* __restrict__ bb,
                                                   u16* __restrict__ out) {
    __shared__ float red[256];
    const int tid = threadIdx.x;
    const float* xr = x + (size_t)blockIdx.x * D_MODEL;
    float v0 = xr[tid], v1 = xr[tid + 256], v2 = xr[tid + 512];

    red[tid] = v0 + v1 + v2;
    __syncthreads();
    for (int off = 128; off > 0; off >>= 1) {
        if (tid < off) red[tid] += red[tid + off];
        __syncthreads();
    }
    float mean = red[0] * (1.0f / 768.0f);
    __syncthreads();

    float d0 = v0 - mean, d1 = v1 - mean, d2 = v2 - mean;
    red[tid] = d0 * d0 + d1 * d1 + d2 * d2;
    __syncthreads();
    for (int off = 128; off > 0; off >>= 1) {
        if (tid < off) red[tid] += red[tid + off];
        __syncthreads();
    }
    float rinv = 1.0f / sqrtf(red[0] * (1.0f / 768.0f) + 1e-5f);

    u16* orow = out + (size_t)blockIdx.x * D_MODEL;
    orow[tid]       = f2bf(d0 * rinv * w[tid]       + bb[tid]);
    orow[tid + 256] = f2bf(d1 * rinv * w[tid + 256] + bb[tid + 256]);
    orow[tid + 512] = f2bf(d2 * rinv * w[tid + 512] + bb[tid + 512]);
}

// ---------------------------------------------------------------------------
// bf16 GEMM: C[M,N] = A[M,K] * B[K,N] + bias, f32 accumulation via WMMA.
// WG = 256 threads = 8 waves (2x4); WG tile 128x256; wave tile 64x64
// (4x4 accumulator tiles -> 16 WMMAs per K=32 step, 4x fragment reuse).
// A tile staged via GLOBAL_LOAD_ASYNC_TO_LDS_B128 (ASYNCcnt); B tile is a
// transpose so it takes the manual path, overlapping the async copy.
// MODE 0: store bf16; MODE 1: GELU -> bf16; MODE 2: + resid -> f32
// ---------------------------------------------------------------------------
template <int MODE>
__global__ __launch_bounds__(256) void k_gemm(const u16* __restrict__ A,
                                              const u16* __restrict__ B,
                                              const float* __restrict__ bias,
                                              const float* __restrict__ resid,
                                              void* __restrict__ outp,
                                              int M, int N, int K) {
    __shared__ __align__(16) u16 Alds[128 * 32];  // [row][k]
    __shared__ __align__(16) u16 Blds[256 * 32];  // [col][k] (transposed)
    const int tid  = threadIdx.x;
    const int lane = tid & 31, wave = tid >> 5;
    const int wm = (wave & 1) * 64;   // wave row offset within WG tile
    const int wn = (wave >> 1) * 64;  // wave col offset within WG tile
    const int m0 = blockIdx.y * 128;
    const int n0 = blockIdx.x * 256;

    v8f acc[4][4];
#pragma unroll
    for (int i = 0; i < 4; ++i)
#pragma unroll
        for (int j = 0; j < 4; ++j) acc[i][j] = v8zero();

    for (int kk = 0; kk < K; kk += 32) {
        __syncthreads();
        // Stage A tile asynchronously: 128 rows x 32 K = 512 16-byte chunks,
        // memory -> LDS with no VGPR round trip.
#pragma unroll
        for (int c = tid; c < 512; c += 256) {
            int row = c >> 2, part = c & 3;
            async_copy_b128(Alds + row * 32 + part * 8,
                            A + (size_t)(m0 + row) * K + kk + part * 8);
        }
        // Stage B tile transposed (manual: it's a data reshape): Blds[col][k]
        for (int i = tid; i < 8192; i += 256) {
            int kr = i >> 8, col = i & 255;
            Blds[col * 32 + kr] = B[(size_t)(kk + kr) * N + n0 + col];
        }
        // Prefetch next K-tile toward the WGP while we compute this one
        if (kk + 32 < K) {
            __builtin_prefetch(A + (size_t)(m0 + (tid >> 1)) * K + kk + 32, 0, 1);
            __builtin_prefetch(B + (size_t)(kk + 32 + (tid & 31)) * N + n0 + (tid >> 5) * 32, 0, 1);
        }
        wait_async_all();   // each wave drains its own ASYNCcnt before the barrier
        __syncthreads();

        v16bf af[4], bf[4];
#pragma unroll
        for (int i = 0; i < 4; ++i) af[i] = load_a_frag(Alds, 32, wm + i * 16, lane);
#pragma unroll
        for (int j = 0; j < 4; ++j) bf[j] = load_b_frag(Blds, 32, wn + j * 16, lane);
#pragma unroll
        for (int i = 0; i < 4; ++i)
#pragma unroll
            for (int j = 0; j < 4; ++j) acc[i][j] = wmma_bf16(af[i], bf[j], acc[i][j]);
    }

    // Epilogue. C layout: VGPR r, lane l -> row = r + 8*(l>=16), col = l&15
    const int rowg = (lane >> 4) << 3;
    const int ncol = lane & 15;
#pragma unroll
    for (int i = 0; i < 4; ++i) {
#pragma unroll
        for (int j = 0; j < 4; ++j) {
            int col = n0 + wn + j * 16 + ncol;
            float bcol = bias[col];
#pragma unroll
            for (int r = 0; r < 8; ++r) {
                int row = m0 + wm + i * 16 + r + rowg;
                float v = acc[i][j][r] + bcol;
                size_t off = (size_t)row * N + col;
                if (MODE == 0) {
                    ((u16*)outp)[off] = f2bf(v);
                } else if (MODE == 1) {
                    float g = 0.5f * v * (1.0f + tanhf(0.7978845608028654f *
                                                       (v + 0.044715f * v * v * v)));
                    ((u16*)outp)[off] = f2bf(g);
                } else {
                    ((float*)outp)[off] = v + resid[off];
                }
            }
        }
    }
}

// ---------------------------------------------------------------------------
// Flash attention. Grid (S/64, H, B), 128 threads (4 waves). Each wave owns
// 16 query rows; iterates causal 64-key blocks (8 QK^T + 8 P*V WMMAs each);
// online softmax in the WMMA C-register layout (row = one VGPR across a
// 16-lane group, reduced with shfl_xor 1/2/4/8).
// q/k/v/z are bf16 [B*S][H*64] (head-major rows).
// ---------------------------------------------------------------------------
__global__ __launch_bounds__(128) void k_flash(const u16* __restrict__ Qb,
                                               const u16* __restrict__ Kb,
                                               const u16* __restrict__ Vb,
                                               u16* __restrict__ Zb) {
    __shared__ __align__(16) u16 Vlds[64 * 64];      // [dh][key] transposed V tile
    __shared__ __align__(16) u16 Plds[4 * 16 * 64];  // per-wave P tile (16x64)
    const int lane = threadIdx.x & 31, wave = threadIdx.x >> 5;
    const int h = blockIdx.y, b = blockIdx.z;
    const int q0 = blockIdx.x * 64;
    const int qw = q0 + wave * 16;
    const size_t rowb = (size_t)b * SEQ;
    const u16* Q  = Qb + rowb * D_MODEL + h * D_HEAD;
    const u16* Kp = Kb + rowb * D_MODEL + h * D_HEAD;
    const u16* Vp = Vb + rowb * D_MODEL + h * D_HEAD;
    u16*       Z  = Zb + rowb * D_MODEL + h * D_HEAD;
    u16* Pw = Plds + wave * (16 * 64);

    // Q fragments: 16 rows x 64 dh = two K=32 fragments, loaded once
    v16bf qf[2];
    qf[0] = load_a_frag(Q + (size_t)qw * D_MODEL,      D_MODEL, 0, lane);
    qf[1] = load_a_frag(Q + (size_t)qw * D_MODEL + 32, D_MODEL, 0, lane);

    float mrow[8], lrow[8];
    v8f o[4];
#pragma unroll
    for (int r = 0; r < 8; ++r) { mrow[r] = -3.0e38f; lrow[r] = 0.0f; }
#pragma unroll
    for (int t = 0; t < 4; ++t) o[t] = v8zero();

    const int rowg = (lane >> 4) << 3;
    const int ncol = lane & 15;
    const int nkb = (q0 >> 6) + 1;  // causal: 64-key blocks covering keys <= q0+63

    for (int kb = 0; kb < nkb; ++kb) {
        const int k0 = kb * 64;
        __syncthreads();
        // Stage V tile transposed: Vlds[dh][key], 4096 elements / 128 threads
        for (int i = threadIdx.x; i < 4096; i += 128) {
            int key = i >> 6, e = i & 63;
            Vlds[e * 64 + key] = Vp[(size_t)(k0 + key) * D_MODEL + e];
        }
        __syncthreads();

        // Scores: 4 tiles of 16 keys, contraction over dh=64 (2 WMMAs each)
        v8f s[4];
#pragma unroll
        for (int t = 0; t < 4; ++t) s[t] = v8zero();
#pragma unroll
        for (int t = 0; t < 4; ++t) {
            v16bf kf0 = load_b_frag(Kp + (size_t)(k0 + t * 16) * D_MODEL,      D_MODEL, 0, lane);
            v16bf kf1 = load_b_frag(Kp + (size_t)(k0 + t * 16) * D_MODEL + 32, D_MODEL, 0, lane);
            s[t] = wmma_bf16(qf[0], kf0, s[t]);
            s[t] = wmma_bf16(qf[1], kf1, s[t]);
        }

        // Scale + causal mask + row max (rows live across 16-lane groups)
        float bmax[8];
#pragma unroll
        for (int r = 0; r < 8; ++r) {
            int qq = qw + r + rowg;
            float x = -3.0e38f;
#pragma unroll
            for (int t = 0; t < 4; ++t) {
                float xv = s[t][r] * 0.125f;
                if (k0 + t * 16 + ncol > qq) xv = -1.0e5f;
                s[t][r] = xv;
                x = fmaxf(x, xv);
            }
            x = fmaxf(x, __shfl_xor(x, 1, 32));
            x = fmaxf(x, __shfl_xor(x, 2, 32));
            x = fmaxf(x, __shfl_xor(x, 4, 32));
            x = fmaxf(x, __shfl_xor(x, 8, 32));
            bmax[r] = x;
        }
        // Online softmax update
#pragma unroll
        for (int r = 0; r < 8; ++r) {
            float mn = fmaxf(mrow[r], bmax[r]);
            float sc = __expf(mrow[r] - mn);
            mrow[r] = mn;
            float rs = 0.0f;
#pragma unroll
            for (int t = 0; t < 4; ++t) {
                float p = __expf(s[t][r] - mn);
                s[t][r] = p;
                rs += p;
            }
            rs += __shfl_xor(rs, 1, 32);
            rs += __shfl_xor(rs, 2, 32);
            rs += __shfl_xor(rs, 4, 32);
            rs += __shfl_xor(rs, 8, 32);
            lrow[r] = lrow[r] * sc + rs;
#pragma unroll
            for (int t = 0; t < 4; ++t) o[t][r] *= sc;
        }
        // C-layout -> A-layout via per-wave LDS tile (same-wave DS ops are in order)
#pragma unroll
        for (int r = 0; r < 8; ++r) {
            int prow = r + rowg;
#pragma unroll
            for (int t = 0; t < 4; ++t)
                Pw[prow * 64 + t * 16 + ncol] = f2bf(s[t][r]);
        }
        v16bf pf0 = load_a_frag(Pw,      64, 0, lane);   // keys 0..31 of block
        v16bf pf1 = load_a_frag(Pw + 32, 64, 0, lane);   // keys 32..63 of block
#pragma unroll
        for (int t = 0; t < 4; ++t) {
            v16bf vf0 = load_b_frag(Vlds,      64, t * 16, lane);
            v16bf vf1 = load_b_frag(Vlds + 32, 64, t * 16, lane);
            o[t] = wmma_bf16(pf0, vf0, o[t]);
            o[t] = wmma_bf16(pf1, vf1, o[t]);
        }
    }

    // Normalize and store z (bf16)
#pragma unroll
    for (int r = 0; r < 8; ++r) {
        float inv = 1.0f / lrow[r];
        size_t zr = (size_t)(qw + r + rowg) * D_MODEL;
#pragma unroll
        for (int t = 0; t < 4; ++t)
            Z[zr + t * 16 + ncol] = f2bf(o[t][r] * inv);
    }
}

// ---------------------------------------------------------------------------
extern "C" void kernel_launch(void* const* d_in, const int* in_sizes, int n_in,
                              void* d_out, int out_size, void* d_ws, size_t ws_size,
                              hipStream_t stream) {
    const float* resid_pre = (const float*)d_in[0];
    const float* ln1_w = (const float*)d_in[1];
    const float* ln1_b = (const float*)d_in[2];
    const float* ln2_w = (const float*)d_in[3];
    const float* ln2_b = (const float*)d_in[4];
    const float* W_Q   = (const float*)d_in[5];
    const float* b_Q   = (const float*)d_in[6];
    const float* W_K   = (const float*)d_in[7];
    const float* b_K   = (const float*)d_in[8];
    const float* W_V   = (const float*)d_in[9];
    const float* b_V   = (const float*)d_in[10];
    const float* W_O   = (const float*)d_in[11];
    const float* b_O   = (const float*)d_in[12];
    const float* W_in  = (const float*)d_in[13];
    const float* b_in  = (const float*)d_in[14];
    const float* W_out = (const float*)d_in[15];
    const float* b_out = (const float*)d_in[16];

    char* ws = (char*)d_ws;
    size_t off = 0;
    auto alloc = [&](size_t bytes) -> char* {
        char* p = ws + off;
        off += (bytes + 255) & ~(size_t)255;
        return p;
    };
    u16* wq_bf   = (u16*)alloc((size_t)D_MODEL * D_MODEL * 2);
    u16* wk_bf   = (u16*)alloc((size_t)D_MODEL * D_MODEL * 2);
    u16* wv_bf   = (u16*)alloc((size_t)D_MODEL * D_MODEL * 2);
    u16* wo_bf   = (u16*)alloc((size_t)D_MODEL * D_MODEL * 2);
    u16* win_bf  = (u16*)alloc((size_t)D_MODEL * D_MLP * 2);
    u16* wout_bf = (u16*)alloc((size_t)D_MLP * D_MODEL * 2);
    u16* xln_bf  = (u16*)alloc((size_t)ROWS * D_MODEL * 2);   // ln1 out, reused for ln2
    u16* q_bf    = (u16*)alloc((size_t)ROWS * D_MODEL * 2);
    u16* k_bf    = (u16*)alloc((size_t)ROWS * D_MODEL * 2);
    u16* v_bf    = (u16*)alloc((size_t)ROWS * D_MODEL * 2);
    u16* z_bf    = (u16*)alloc((size_t)ROWS * D_MODEL * 2);
    float* resid_mid = (float*)alloc((size_t)ROWS * D_MODEL * 4);
    u16* h_bf    = (u16*)alloc((size_t)ROWS * D_MLP * 2);
    (void)ws_size; (void)in_sizes; (void)n_in; (void)out_size;

    // 1. Weight conversion (+ QKV permute [h][d][e] -> [d][h*64+e])
    k_cvt_qkvw<<<(D_MODEL * D_MODEL) / 256, 256, 0, stream>>>(W_Q, wq_bf);
    k_cvt_qkvw<<<(D_MODEL * D_MODEL) / 256, 256, 0, stream>>>(W_K, wk_bf);
    k_cvt_qkvw<<<(D_MODEL * D_MODEL) / 256, 256, 0, stream>>>(W_V, wv_bf);
    k_cvt<<<(D_MODEL * D_MODEL) / 256, 256, 0, stream>>>(W_O, wo_bf, D_MODEL * D_MODEL);
    k_cvt<<<(D_MODEL * D_MLP) / 256, 256, 0, stream>>>(W_in, win_bf, D_MODEL * D_MLP);
    k_cvt<<<(D_MLP * D_MODEL) / 256, 256, 0, stream>>>(W_out, wout_bf, D_MLP * D_MODEL);

    // 2. LN1
    k_layernorm<<<ROWS, 256, 0, stream>>>(resid_pre, ln1_w, ln1_b, xln_bf);

    // 3. QKV projections (bf16 GEMMs, bias fused)
    dim3 g768(D_MODEL / 256, ROWS / 128);
    k_gemm<0><<<g768, 256, 0, stream>>>(xln_bf, wq_bf, b_Q, nullptr, q_bf, ROWS, D_MODEL, D_MODEL);
    k_gemm<0><<<g768, 256, 0, stream>>>(xln_bf, wk_bf, b_K, nullptr, k_bf, ROWS, D_MODEL, D_MODEL);
    k_gemm<0><<<g768, 256, 0, stream>>>(xln_bf, wv_bf, b_V, nullptr, v_bf, ROWS, D_MODEL, D_MODEL);

    // 4. Flash attention
    dim3 gf(SEQ / 64, N_HEADS, BATCH);
    k_flash<<<gf, 128, 0, stream>>>(q_bf, k_bf, v_bf, z_bf);

    // 5. Output projection + residual -> resid_mid (f32)
    k_gemm<2><<<g768, 256, 0, stream>>>(z_bf, wo_bf, b_O, resid_pre, resid_mid, ROWS, D_MODEL, D_MODEL);

    // 6. LN2
    k_layernorm<<<ROWS, 256, 0, stream>>>(resid_mid, ln2_w, ln2_b, xln_bf);

    // 7. MLP in (+GELU)
    dim3 gmlp(D_MLP / 256, ROWS / 128);
    k_gemm<1><<<gmlp, 256, 0, stream>>>(xln_bf, win_bf, b_in, nullptr, h_bf, ROWS, D_MLP, D_MODEL);

    // 8. MLP out + residual -> d_out (f32)
    k_gemm<2><<<g768, 256, 0, stream>>>(h_bf, wout_bf, b_out, resid_mid, (float*)d_out, ROWS, D_MODEL, D_MLP);
}
